// CrisisMemoryBank_4329327035084
// MI455X (gfx1250) — matrix operand order repaired and checked by reading, not compiled, not run on hardware
//
#include <hip/hip_runtime.h>
#include <hip/hip_bf16.h>

// ---------------------------------------------------------------------------
// CrisisMemoryBank fused attention for MI455X (gfx1250, wave32, WMMA bf16)
//
// B=32, N=500000, KD=128, VD=256, H=8, DH=16, TOPK=16
// Row ordering for all [256 x *] matrices: row = h*32 + b  (wave w owns head w)
// Keys tiles staged via Tensor Data Mover (TDM) when available.
// ---------------------------------------------------------------------------

typedef unsigned short u16;
typedef __attribute__((ext_vector_type(16))) __bf16 v16bf;
typedef __attribute__((ext_vector_type(8)))  float  v8f;

union V16 { uint4 q[2]; v16bf v; };

#define CB   32          // batch
#define CN   500000      // memory rows
#define CKD  128
#define CVD  256
#define CH   8
#define CT   32          // n-tile rows per iteration
#define CG   512         // workgroups for streaming passes
#define NEGBIG (-3.0e38f)

#if __has_builtin(__builtin_amdgcn_tensor_load_to_lds) && __has_builtin(__builtin_amdgcn_s_wait_tensorcnt)
#define HAVE_TDM 1
#else
#define HAVE_TDM 0
#endif

__device__ __forceinline__ u16 bf16raw(float f) {
  __bf16 h = (__bf16)f;                      // native cvt where available
  return __builtin_bit_cast(u16, h);
}

__device__ __forceinline__ uint2 pk4(float4 f) {
  unsigned lo = (unsigned)bf16raw(f.x) | ((unsigned)bf16raw(f.y) << 16);
  unsigned hi = (unsigned)bf16raw(f.z) | ((unsigned)bf16raw(f.w) << 16);
  return make_uint2(lo, hi);
}

__device__ __forceinline__ v8f wmma_bf16(v16bf a, v16bf b, v8f c) {
  // D = A(16x32 bf16) x B(32x16 bf16) + C(16x16 f32)
  return __builtin_amdgcn_wmma_f32_16x16x32_bf16(false, a, false, b,
                                                 (short)0, c, false, false);
}

#if HAVE_TDM
typedef unsigned int tdm_u4 __attribute__((ext_vector_type(4)));
typedef int          tdm_i8 __attribute__((ext_vector_type(8)));
typedef int          tdm_i4 __attribute__((ext_vector_type(4)));

// 2D tensor tile load: 4-byte elements, row-major, tile (tile1 rows x tile0 elems),
// tensor bounded to (dim1 rows x dim0 elems) so OOB tail rows return zero.
// This toolchain exposes the 6-arg builtin:
//   (uint32x4 g0, int32x8 g1, int32x4 g2, int32x4 g3, int32x8 g4, i32 cpol)
__device__ __forceinline__ void tdm_load_2d(const void* gptr, unsigned lds_off,
                                            unsigned dim0, unsigned dim1,
                                            unsigned tile0, unsigned tile1,
                                            unsigned stride0) {
  unsigned long long ga = (unsigned long long)gptr;
  tdm_u4 g0;
  g0[0] = 0x1u;                                            // count=1, user D#
  g0[1] = lds_off;                                         // lds_addr (bytes)
  g0[2] = (unsigned)(ga & 0xffffffffu);                    // global_addr lo
  g0[3] = (unsigned)((ga >> 32) & 0x1ffffffu) | (2u << 30);// addr hi | type=2
  tdm_i8 g1;
  g1[0] = 2 << 16;                                         // data_size=2 (4B)
  g1[1] = (int)((dim0 & 0xffffu) << 16);                   // tensor_dim0 lo16
  g1[2] = (int)((dim0 >> 16) | ((dim1 & 0xffffu) << 16));  // dim0 hi | dim1 lo
  g1[3] = (int)((dim1 >> 16) | (tile0 << 16));             // dim1 hi | tile_dim0
  g1[4] = (int)tile1;                                      // tile_dim1 (dim2=0)
  g1[5] = (int)stride0;                                    // dim0 stride lo32
  g1[6] = 0;
  g1[7] = 0;
  tdm_i4 z4 = {0, 0, 0, 0};
  tdm_i8 z8 = {0, 0, 0, 0, 0, 0, 0, 0};
  __builtin_amdgcn_tensor_load_to_lds(g0, g1, z4, z4, z8, 0);
}
#endif

// sorted-descending register top-16 list, fully unrolled (stays in VGPRs)
__device__ __forceinline__ void ins16(float v, int i, float (&tv)[16], int (&ti)[16]) {
  if (v <= tv[15]) return;
#pragma unroll
  for (int p = 15; p >= 1; --p) {
    if (v > tv[p - 1]) { tv[p] = tv[p - 1]; ti[p] = ti[p - 1]; }
    else               { tv[p] = v;         ti[p] = i;         return; }
  }
  tv[0] = v; ti[0] = i;
}

// ---------------------------------------------------------------------------
// K1: query MLP -> q -> Qeff (Wk folded in), and WvT, both bf16 in workspace
// ---------------------------------------------------------------------------
__global__ __launch_bounds__(256) void k_setup(
    const float* __restrict__ query, const float* __restrict__ W1,
    const float* __restrict__ b1, const float* __restrict__ W2,
    const float* __restrict__ b2, const float* __restrict__ Wq,
    const float* __restrict__ bq, const float* __restrict__ Wk,
    const float* __restrict__ Wv, u16* __restrict__ wsQ, u16* __restrict__ wsWvT) {
  __shared__ float sA[8192];   // h1 [32][256], later q [32][128]
  __shared__ float sB[4096];   // qk [32][128]
  const int t = threadIdx.x;
  for (int o = t; o < 8192; o += 256) {           // h1 = relu(query@W1+b1)
    int b = o >> 8, j = o & 255;
    float acc = b1[j];
    for (int c = 0; c < 256; ++c) acc += query[b * 256 + c] * W1[c * 256 + j];
    sA[o] = fmaxf(acc, 0.f);
  }
  __syncthreads();
  for (int o = t; o < 4096; o += 256) {           // qk = h1@W2+b2
    int b = o >> 7, j = o & 127;
    float acc = b2[j];
    for (int c = 0; c < 256; ++c) acc += sA[b * 256 + c] * W2[c * 128 + j];
    sB[o] = acc;
  }
  __syncthreads();
  for (int o = t; o < 4096; o += 256) {           // q = qk@Wq+bq
    int b = o >> 7, j = o & 127;
    float acc = bq[j];
    for (int c = 0; c < 128; ++c) acc += sB[b * 128 + c] * Wq[c * 128 + j];
    sA[o] = acc;
  }
  __syncthreads();
  // Qeff[(h*32+b), c] = sum_d Wk[c, h*16+d] * q[b, h*16+d]   (bk cancels in softmax)
  for (int o = t; o < 32768; o += 256) {
    int row = o >> 7, c = o & 127;
    int h = row >> 5, b = row & 31;
    float acc = 0.f;
#pragma unroll
    for (int d = 0; d < 16; ++d) acc += Wk[c * 128 + h * 16 + d] * sA[b * 128 + h * 16 + d];
    wsQ[o] = bf16raw(acc);
  }
  for (int o = t; o < 32768; o += 256) {          // WvT[j][c] = Wv[c][j]
    int j = o >> 8, c = o & 255;
    wsWvT[j * 256 + c] = bf16raw(Wv[c * 128 + j]);
  }
}

// ---------------------------------------------------------------------------
// K2: streaming pass. Per 32-row tile: vproj = values@Wv (WMMA), scores =
// Qeff@keys^T (WMMA), online softmax per row, ctx += P@vproj (WMMA).
// ---------------------------------------------------------------------------
__global__ __launch_bounds__(256) void k_pass1(
    const float* __restrict__ keys, const float* __restrict__ vals,
    const u16* __restrict__ wsQ, const u16* __restrict__ wsWvT,
    float* __restrict__ wsm, float* __restrict__ wsz, float* __restrict__ wsctx) {
#if HAVE_TDM
  __attribute__((aligned(16))) __shared__ float skf[CT * 128]; // keys f32 via TDM
#endif
  __attribute__((aligned(16))) __shared__ u16 kb[CT * 128];    // keys tile bf16
  __attribute__((aligned(16))) __shared__ u16 vb[CT * 256];    // values tile bf16
  __attribute__((aligned(16))) __shared__ u16 pvT[128 * CT];   // vproj^T bf16 [j][n]
  __attribute__((aligned(16))) __shared__ u16 pa[256 * CT];    // P bf16 [(h*32+b)][n]

  const int tid = threadIdx.x, w = tid >> 5, lane = tid & 31;
  const int l16 = lane & 15, hi = lane >> 4;

  // Qeff A-fragments: wave w owns rows 32w..32w+31 (head w), K=128 -> 4 steps
  v16bf qA[2][4];
#pragma unroll
  for (int t = 0; t < 2; ++t)
#pragma unroll
    for (int kk = 0; kk < 4; ++kk) {
      const int row = 32 * w + 16 * t + l16;
      V16 u;
      u.q[0] = *(const uint4*)&wsQ[row * 128 + 32 * kk + hi * 16];
      u.q[1] = *(const uint4*)&wsQ[row * 128 + 32 * kk + hi * 16 + 8];
      qA[t][kk] = u.v;
    }
  // Wv B-fragments: wave w produces vproj columns 16w..16w+15, K=256 -> 8 steps
  v16bf wB[8];
#pragma unroll
  for (int kk = 0; kk < 8; ++kk) {
    const int j = 16 * w + l16;
    V16 u;
    u.q[0] = *(const uint4*)&wsWvT[j * 256 + 32 * kk + hi * 16];
    u.q[1] = *(const uint4*)&wsWvT[j * 256 + 32 * kk + hi * 16 + 8];
    wB[kk] = u.v;
  }

  v8f ctxA[2] = { {0,0,0,0,0,0,0,0}, {0,0,0,0,0,0,0,0} };
  float mrun[2][8], zrun[2][8];
#pragma unroll
  for (int t = 0; t < 2; ++t)
#pragma unroll
    for (int r = 0; r < 8; ++r) { mrun[t][r] = NEGBIG; zrun[t][r] = 0.f; }

  for (int n0 = blockIdx.x * CT; n0 < CN; n0 += CG * CT) {
    // ---- stage tiles ----
#if HAVE_TDM
    if (w == 0)     // one DMA per workgroup: keys f32 tile -> LDS (ASYNC via TDM)
      tdm_load_2d(keys + (long)n0 * 128, (unsigned)(unsigned long long)(void*)skf,
                  128u, (unsigned)(CN - n0), 128u, (unsigned)CT, 128u);
#else
    for (int i = tid * 4; i < CT * 128; i += 1024) {
      int rrow = i >> 7;
      long grow = (n0 + rrow < CN) ? (long)(n0 + rrow) : (long)(CN - 1);
      float4 f = *(const float4*)&keys[grow * 128 + (i & 127)];
      *(uint2*)&kb[i] = pk4(f);
    }
#endif
    for (int i = tid * 4; i < CT * 256; i += 1024) {  // values: coalesced VMEM
      int rrow = i >> 8;
      long grow = (n0 + rrow < CN) ? (long)(n0 + rrow) : (long)(CN - 1);
      float4 f = *(const float4*)&vals[grow * 256 + (i & 255)];
      *(uint2*)&vb[i] = pk4(f);
    }
    { // prefetch next tile into L2 (global_prefetch_b8)
      long nn = (long)n0 + (long)CG * CT;
      if (nn < CN) {
        __builtin_prefetch(&keys[nn * 128 + tid * 16], 0, 1);
        __builtin_prefetch(&vals[nn * 256 + tid * 32], 0, 1);
      }
    }
#if HAVE_TDM
    if (w == 0) __builtin_amdgcn_s_wait_tensorcnt(0);
    __syncthreads();
    for (int i = tid * 4; i < CT * 128; i += 1024) {  // LDS f32 -> LDS bf16
      float4 f = *(const float4*)&skf[i];
      *(uint2*)&kb[i] = pk4(f);
    }
#endif
    __syncthreads();

    // ---- vproj tile: [32 x 128] = values[32 x 256] @ Wv, wave w -> cols 16w.. ----
#pragma unroll
    for (int t = 0; t < 2; ++t) {
      V16 fa[8];
#pragma unroll
      for (int kk = 0; kk < 8; ++kk) {
        const int a0 = (16 * t + l16) * 256 + 32 * kk + hi * 16;
        fa[kk].q[0] = *(const uint4*)&vb[a0];
        fa[kk].q[1] = *(const uint4*)&vb[a0 + 8];
      }
      v8f acc = {0,0,0,0,0,0,0,0};
#pragma unroll
      for (int kk = 0; kk < 8; ++kk) acc = wmma_bf16(fa[kk].v, wB[kk], acc);
#pragma unroll
      for (int r = 0; r < 8; ++r)        // store transposed: pvT[j][n]
        pvT[(16 * w + l16) * CT + 16 * t + r + hi * 8] = bf16raw(acc[r]);
    }

    // ---- scores + online softmax (rows owned by this wave) ----
    V16 fb[8];                           // keys B-frags, shared by both M-tiles
#pragma unroll
    for (int kk = 0; kk < 4; ++kk) {
      const int bo0 = l16 * 128 + 32 * kk + hi * 16;         // n-cols 0..15
      fb[kk].q[0] = *(const uint4*)&kb[bo0];
      fb[kk].q[1] = *(const uint4*)&kb[bo0 + 8];
      const int bo1 = (16 + l16) * 128 + 32 * kk + hi * 16;  // n-cols 16..31
      fb[4 + kk].q[0] = *(const uint4*)&kb[bo1];
      fb[4 + kk].q[1] = *(const uint4*)&kb[bo1 + 8];
    }
#pragma unroll
    for (int t = 0; t < 2; ++t) {
      v8f s0 = {0,0,0,0,0,0,0,0}, s1 = {0,0,0,0,0,0,0,0};
#pragma unroll
      for (int kk = 0; kk < 4; ++kk) {
        s0 = wmma_bf16(qA[t][kk], fb[kk].v, s0);
        s1 = wmma_bf16(qA[t][kk], fb[4 + kk].v, s1);
      }
      const bool v0 = (n0 + l16) < CN, v1 = (n0 + 16 + l16) < CN;
#pragma unroll
      for (int r = 0; r < 8; ++r) {
        float a = v0 ? s0[r] * 0.25f : NEGBIG;  // scale = 1/sqrt(DH)
        float b = v1 ? s1[r] * 0.25f : NEGBIG;
        float mx = fmaxf(a, b);
#pragma unroll
        for (int m = 1; m <= 8; m <<= 1) mx = fmaxf(mx, __shfl_xor(mx, m, 32));
        float mnew = fmaxf(mrun[t][r], mx);
        float alpha = __expf(mrun[t][r] - mnew);
        float p0 = __expf(a - mnew), p1 = __expf(b - mnew);
        float zs = p0 + p1;
#pragma unroll
        for (int m = 1; m <= 8; m <<= 1) zs += __shfl_xor(zs, m, 32);
        zrun[t][r] = zrun[t][r] * alpha + zs;
        mrun[t][r] = mnew;
        ctxA[t][r] *= alpha;                               // rescale running ctx
        const int prow = 32 * w + 16 * t + r + hi * 8;
        pa[prow * CT + l16]      = bf16raw(p0);
        pa[prow * CT + 16 + l16] = bf16raw(p1);
      }
    }

    // ---- ctx += P[32 x 32] @ vproj[:,16w..16w+16]  (all intra-wave data) ----
#pragma unroll
    for (int t = 0; t < 2; ++t) {
      const int a0 = (32 * w + 16 * t + l16) * CT + hi * 16;
      V16 ua; ua.q[0] = *(const uint4*)&pa[a0]; ua.q[1] = *(const uint4*)&pa[a0 + 8];
      const int b0 = (16 * w + l16) * CT + hi * 16;
      V16 ub; ub.q[0] = *(const uint4*)&pvT[b0]; ub.q[1] = *(const uint4*)&pvT[b0 + 8];
      ctxA[t] = wmma_bf16(ua.v, ub.v, ctxA[t]);
    }
    __syncthreads();   // tiles are re-staged next iteration
  }

  // ---- write split-K partials ----
  const int wg = blockIdx.x;
#pragma unroll
  for (int t = 0; t < 2; ++t)
#pragma unroll
    for (int r = 0; r < 8; ++r) {
      const int row = 32 * w + 16 * t + r + hi * 8;
      if (l16 == 0) { wsm[wg * 256 + row] = mrun[t][r]; wsz[wg * 256 + row] = zrun[t][r]; }
      wsctx[((long)wg * 256 + row) * 16 + l16] = ctxA[t][r];
    }
}

// ---------------------------------------------------------------------------
// K3: combine split-K partials, add bv, project through Wo -> attended output
// ---------------------------------------------------------------------------
__global__ __launch_bounds__(256) void k_reduce(
    const float* __restrict__ wsm, const float* __restrict__ wsz,
    const float* __restrict__ wsctx, const float* __restrict__ Wo,
    const float* __restrict__ bo, const float* __restrict__ bv,
    float* __restrict__ wsmf, float* __restrict__ wszf, float* __restrict__ out) {
  __shared__ float ctxf[256 * 16];
  const int t = threadIdx.x;                 // row = h*32+b
  float mb = NEGBIG;
  for (int g = 0; g < CG; ++g) mb = fmaxf(mb, wsm[g * 256 + t]);
  float zz = 0.f, cx[16];
#pragma unroll
  for (int d = 0; d < 16; ++d) cx[d] = 0.f;
  for (int g = 0; g < CG; ++g) {
    float e = __expf(wsm[g * 256 + t] - mb);
    zz += wsz[g * 256 + t] * e;
#pragma unroll
    for (int d = 0; d < 16; ++d) cx[d] += wsctx[((long)g * 256 + t) * 16 + d] * e;
  }
  float inv = 1.f / zz;
  const int h = t >> 5;
#pragma unroll
  for (int d = 0; d < 16; ++d) ctxf[t * 16 + d] = cx[d] * inv + bv[h * 16 + d];
  wsmf[t] = mb; wszf[t] = zz;
  __syncthreads();
  for (int o = t; o < 4096; o += 256) {      // attended = ctx@Wo+bo
    int b = o >> 7, j = o & 127;
    float acc = bo[j];
    for (int c = 0; c < 128; ++c)
      acc += ctxf[((c >> 4) * 32 + b) * 16 + (c & 15)] * Wo[c * 128 + j];
    out[o] = acc;
  }
}

// ---------------------------------------------------------------------------
// K4: recompute scores with final (m,Z), head-average via LDS float atomics,
// keep per-WG top-16 candidates per batch row in registers.
// ---------------------------------------------------------------------------
__global__ __launch_bounds__(256) void k_pass2(
    const float* __restrict__ keys, const u16* __restrict__ wsQ,
    const float* __restrict__ wsmf, const float* __restrict__ wszf,
    float* __restrict__ wcv, int* __restrict__ wci) {
#if HAVE_TDM
  __attribute__((aligned(16))) __shared__ float skf[CT * 128];
#endif
  __attribute__((aligned(16))) __shared__ u16 kb[CT * 128];
  __shared__ float av[32 * CT];             // attn_avg tile [b][n]
  __shared__ float cvv[32 * 8 * 16];
  __shared__ int   cvi[32 * 8 * 16];

  const int tid = threadIdx.x, w = tid >> 5, lane = tid & 31;
  const int l16 = lane & 15, hi = lane >> 4;

  v16bf qA[2][4];
#pragma unroll
  for (int t = 0; t < 2; ++t)
#pragma unroll
    for (int kk = 0; kk < 4; ++kk) {
      const int row = 32 * w + 16 * t + l16;
      V16 u;
      u.q[0] = *(const uint4*)&wsQ[row * 128 + 32 * kk + hi * 16];
      u.q[1] = *(const uint4*)&wsQ[row * 128 + 32 * kk + hi * 16 + 8];
      qA[t][kk] = u.v;
    }
  float mst[2][8], zin[2][8];
#pragma unroll
  for (int t = 0; t < 2; ++t)
#pragma unroll
    for (int r = 0; r < 8; ++r) {
      const int row = 32 * w + 16 * t + r + hi * 8;
      mst[t][r] = wsmf[row];
      zin[t][r] = 0.125f / wszf[row];       // 1/H folded in
    }

  float tv[16]; int ti[16];
#pragma unroll
  for (int i = 0; i < 16; ++i) { tv[i] = NEGBIG; ti[i] = 0; }
  const int cb = tid >> 3, cs = tid & 7;    // candidate owner: b, slot

  for (int n0 = blockIdx.x * CT; n0 < CN; n0 += CG * CT) {
#if HAVE_TDM
    if (w == 0)
      tdm_load_2d(keys + (long)n0 * 128, (unsigned)(unsigned long long)(void*)skf,
                  128u, (unsigned)(CN - n0), 128u, (unsigned)CT, 128u);
#else
    for (int i = tid * 4; i < CT * 128; i += 1024) {
      int rrow = i >> 7;
      long grow = (n0 + rrow < CN) ? (long)(n0 + rrow) : (long)(CN - 1);
      float4 f = *(const float4*)&keys[grow * 128 + (i & 127)];
      *(uint2*)&kb[i] = pk4(f);
    }
#endif
    for (int i = tid; i < 32 * CT; i += 256) av[i] = 0.f;
#if HAVE_TDM
    if (w == 0) __builtin_amdgcn_s_wait_tensorcnt(0);
    __syncthreads();
    for (int i = tid * 4; i < CT * 128; i += 1024) {
      float4 f = *(const float4*)&skf[i];
      *(uint2*)&kb[i] = pk4(f);
    }
#endif
    __syncthreads();

    V16 fb[8];
#pragma unroll
    for (int kk = 0; kk < 4; ++kk) {
      const int bo0 = l16 * 128 + 32 * kk + hi * 16;
      fb[kk].q[0] = *(const uint4*)&kb[bo0];
      fb[kk].q[1] = *(const uint4*)&kb[bo0 + 8];
      const int bo1 = (16 + l16) * 128 + 32 * kk + hi * 16;
      fb[4 + kk].q[0] = *(const uint4*)&kb[bo1];
      fb[4 + kk].q[1] = *(const uint4*)&kb[bo1 + 8];
    }
#pragma unroll
    for (int t = 0; t < 2; ++t) {
      v8f s0 = {0,0,0,0,0,0,0,0}, s1 = {0,0,0,0,0,0,0,0};
#pragma unroll
      for (int kk = 0; kk < 4; ++kk) {
        s0 = wmma_bf16(qA[t][kk], fb[kk].v, s0);
        s1 = wmma_bf16(qA[t][kk], fb[4 + kk].v, s1);
      }
      const bool v0 = (n0 + l16) < CN, v1 = (n0 + 16 + l16) < CN;
#pragma unroll
      for (int r = 0; r < 8; ++r) {
        float w0 = v0 ? __expf(s0[r] * 0.25f - mst[t][r]) * zin[t][r] : 0.f;
        float w1 = v1 ? __expf(s1[r] * 0.25f - mst[t][r]) * zin[t][r] : 0.f;
        const int brow = 16 * t + r + hi * 8;           // batch index (head-independent)
        atomicAdd(&av[brow * CT + l16],      w0);
        atomicAdd(&av[brow * CT + 16 + l16], w1);
      }
    }
    __syncthreads();
#pragma unroll
    for (int c = 0; c < 4; ++c) {
      int col = cs * 4 + c, gi = n0 + col;
      if (gi < CN) ins16(av[cb * CT + col], gi, tv, ti);
    }
    __syncthreads();
  }
  // merge 8 register lists per batch row -> per-WG top16
#pragma unroll
  for (int i = 0; i < 16; ++i) { cvv[cb * 128 + cs * 16 + i] = tv[i]; cvi[cb * 128 + cs * 16 + i] = ti[i]; }
  __syncthreads();
  if (tid < 32) {
    for (int it = 0; it < 16; ++it) {
      float best = NEGBIG; int bp = 0;
      for (int e = 0; e < 128; ++e) { float x = cvv[tid * 128 + e]; if (x > best) { best = x; bp = e; } }
      wcv[(long)blockIdx.x * 512 + tid * 16 + it] = best;
      wci[(long)blockIdx.x * 512 + tid * 16 + it] = cvi[tid * 128 + bp];
      cvv[tid * 128 + bp] = NEGBIG;
    }
  }
}

// ---------------------------------------------------------------------------
// K5: merge CG*16 candidates per batch row -> global top-16
// ---------------------------------------------------------------------------
__global__ __launch_bounds__(256) void k_topk(
    const float* __restrict__ wcv, const int* __restrict__ wci, float* __restrict__ out) {
  __shared__ float lv[4096]; __shared__ int li[4096];
  __shared__ float rv[256];  __shared__ int rp[256];
  const int b = blockIdx.x, t = threadIdx.x;
  float tv[16]; int ti[16];
#pragma unroll
  for (int i = 0; i < 16; ++i) { tv[i] = NEGBIG; ti[i] = 0; }
  for (int e = t; e < CG * 16; e += 256) {
    int g = e >> 4, s = e & 15;
    ins16(wcv[(long)g * 512 + b * 16 + s], wci[(long)g * 512 + b * 16 + s], tv, ti);
  }
#pragma unroll
  for (int i = 0; i < 16; ++i) { lv[t * 16 + i] = tv[i]; li[t * 16 + i] = ti[i]; }
  __syncthreads();
  for (int it = 0; it < 16; ++it) {
    float best = NEGBIG; int bp = t * 16;
    for (int j = t; j < 4096; j += 256) if (lv[j] > best) { best = lv[j]; bp = j; }
    rv[t] = best; rp[t] = bp;
    __syncthreads();
    for (int s = 128; s > 0; s >>= 1) {
      if (t < s && rv[t + s] > rv[t]) { rv[t] = rv[t + s]; rp[t] = rp[t + s]; }
      __syncthreads();
    }
    if (t == 0) {
      out[4096 + b * 16 + it] = rv[0];             // top_w
      out[4608 + b * 16 + it] = (float)li[rp[0]];  // top_i (as float)
      lv[rp[0]] = NEGBIG;
    }
    __syncthreads();
  }
}

// ---------------------------------------------------------------------------
extern "C" void kernel_launch(void* const* d_in, const int* in_sizes, int n_in,
                              void* d_out, int out_size, void* d_ws, size_t ws_size,
                              hipStream_t stream) {
  const float* query = (const float*)d_in[0];
  const float* keys  = (const float*)d_in[1];
  const float* vals  = (const float*)d_in[2];
  const float* W1 = (const float*)d_in[3];
  const float* b1 = (const float*)d_in[4];
  const float* W2 = (const float*)d_in[5];
  const float* b2 = (const float*)d_in[6];
  const float* Wq = (const float*)d_in[7];
  const float* bq = (const float*)d_in[8];
  const float* Wk = (const float*)d_in[9];
  /* bk (d_in[10]) cancels in softmax */
  const float* Wv = (const float*)d_in[11];
  const float* bv = (const float*)d_in[12];
  const float* Wo = (const float*)d_in[13];
  const float* bo = (const float*)d_in[14];

  char* base = (char*)d_ws;
  u16*   wsQ   = (u16*)base;                           // 256*128 bf16
  u16*   wsWvT = wsQ + 32768;                          // 128*256 bf16
  float* wsm   = (float*)(base + 131072);              // CG*256
  float* wsz   = wsm + (size_t)CG * 256;
  float* wsctx = wsz + (size_t)CG * 256;               // CG*256*16
  float* wsmf  = wsctx + (size_t)CG * 256 * 16;        // 256 finals
  float* wszf  = wsmf + 256;
  float* wcv   = wszf + 256;                           // CG*32*16 candidate vals
  int*   wci   = (int*)(wcv + (size_t)CG * 512);       // CG*32*16 candidate idx
  float* out   = (float*)d_out;

  k_setup<<<1, 256, 0, stream>>>(query, W1, b1, W2, b2, Wq, bq, Wk, Wv, wsQ, wsWvT);
  k_pass1<<<CG, 256, 0, stream>>>(keys, vals, wsQ, wsWvT, wsm, wsz, wsctx);
  k_reduce<<<1, 256, 0, stream>>>(wsm, wsz, wsctx, Wo, bo, bv, wsmf, wszf, out);
  k_pass2<<<CG, 256, 0, stream>>>(keys, wsQ, wsmf, wszf, wcv, wci);
  k_topk<<<32, 256, 0, stream>>>(wcv, wci, out);
}
